// Glm4MoeNaiveMoeHybrid_29231547416690
// MI455X (gfx1250) — matrix operand order repaired and checked by reading, not compiled, not run on hardware
//
#include <hip/hip_runtime.h>
#include <hip/hip_bf16.h>
#include <math.h>

// Problem constants (from reference): T,D,I,E,K
#define T_TOK 2048
#define D_DIM 2048
#define I_DIM 1408
#define E_EXP 16
#define K_TOP 4

typedef __attribute__((ext_vector_type(16))) __bf16 v16bf;
typedef __attribute__((ext_vector_type(8)))  float  v8f;
typedef __attribute__((ext_vector_type(4)))  float  f32x4;
typedef __attribute__((ext_vector_type(4)))  unsigned int u32x4;

// Fragment viewed either as WMMA operand (v16bf) or as two 16B chunks for LDS loads.
union BfFrag {
    v16bf v;
    f32x4 q[2];
};

__device__ __forceinline__ void st_bf16x4(__bf16* dst, f32x4 f) {
    dst[0] = (__bf16)f.x; dst[1] = (__bf16)f.y;
    dst[2] = (__bf16)f.z; dst[3] = (__bf16)f.w;
}

// CDNA5 async global->LDS copy of 16 bytes (per-lane addresses, ASYNCcnt-tracked).
// lds_off: 32-bit LDS byte address VGPR; gsrc: 64-bit global address.
__device__ __forceinline__ void async_copy_b128(unsigned lds_off, const void* gsrc) {
    asm volatile("global_load_async_to_lds_b128 %0, %1, off"
                 :: "v"(lds_off), "v"((unsigned long long)(size_t)gsrc)
                 : "memory");
}

__device__ __forceinline__ void wait_asynccnt0() {
#if __has_builtin(__builtin_amdgcn_s_wait_asynccnt)
    __builtin_amdgcn_s_wait_asynccnt(0);
#else
    asm volatile("s_wait_asynccnt 0" ::: "memory");
#endif
}

// ---------------------------------------------------------------------------
// Kernel 0: zero output accumulator + expert counters
// ---------------------------------------------------------------------------
__global__ __launch_bounds__(256) void moe_zero_kernel(float* __restrict__ out, int n4,
                                                       int* __restrict__ counts) {
    int i = blockIdx.x * blockDim.x + threadIdx.x;
    if (i < n4) {
        f32x4 z = {0.f, 0.f, 0.f, 0.f};
        ((f32x4*)out)[i] = z;
    }
    if (blockIdx.x == 0 && threadIdx.x < E_EXP) counts[threadIdx.x] = 0;
}

// ---------------------------------------------------------------------------
// Kernel 1: deterministic routing — build per-expert token lists (no atomics).
// ---------------------------------------------------------------------------
__global__ void moe_route_kernel(const int* __restrict__ topk,
                                 const float* __restrict__ topw,
                                 int* __restrict__ counts,
                                 int* __restrict__ list,
                                 float* __restrict__ wlist) {
    if (blockIdx.x != 0 || threadIdx.x != 0) return;
    for (int t = 0; t < T_TOK; ++t) {
        for (int k = 0; k < K_TOP; ++k) {
            int e = topk[t * K_TOP + k];
            if (e < 0 || e >= E_EXP) continue;
            int p = counts[e];
            if (p < T_TOK) {
                list[e * T_TOK + p]  = t;
                wlist[e * T_TOK + p] = topw[t * K_TOP + k];
                counts[e] = p + 1;
            }
        }
    }
}

// ---------------------------------------------------------------------------
// Kernel 2: gate_up GEMM + fused SiLU*up.
// Block: 256 threads (8 waves). Tile: 128 gathered tokens x 64 cols of I,
// K-step 32 over D. Each wave owns 16 rows, 4 N-subtiles, gate+up accums.
// ---------------------------------------------------------------------------
__global__ __launch_bounds__(256) void moe_gateup_kernel(
    const float* __restrict__ hs,      // (T, D) fp32
    const float* __restrict__ w_gu,    // (E, 2I, D) fp32
    const int*   __restrict__ counts,
    const int*   __restrict__ list,
    __bf16*      __restrict__ hbuf)    // (E*T, I) bf16 gathered activations
{
    const int e     = blockIdx.z;
    const int mtile = blockIdx.y;      // 128-row tiles over gathered tokens
    const int ntile = blockIdx.x;      // 64-col tiles over I (22 tiles)
    const int cnt   = counts[e];
    if (mtile * 128 >= cnt) return;

    __shared__ __align__(16) __bf16 lA[128 * 32];
    __shared__ __align__(16) __bf16 lG[64 * 32];
    __shared__ __align__(16) __bf16 lU[64 * 32];

    const int tid  = threadIdx.x;
    const int lane = tid & 31;
    const int wave = tid >> 5;

    const float* wg = w_gu + (size_t)e * (2 * I_DIM) * D_DIM + (size_t)(ntile * 64) * D_DIM;
    const float* wu = wg + (size_t)I_DIM * D_DIM;

    // Hoisted gather state: token per A-tile row this thread services (loop-invariant).
    const int arow = tid >> 3;            // 0..31 (4 passes of 32 rows)
    const int acol = (tid & 7) * 4;       // 0,4,...,28
    const float* asrc[4];
    #pragma unroll
    for (int p = 0; p < 4; ++p) {
        int g = mtile * 128 + p * 32 + arow;
        asrc[p] = (g < cnt) ? (hs + (size_t)list[e * T_TOK + g] * D_DIM + acol) : nullptr;
    }

    v8f accg[4] = {};
    v8f accu[4] = {};

    for (int k0 = 0; k0 < D_DIM; k0 += 32) {
        __syncthreads();
        // --- A tile: 128 rows x 32 cols, gather + fp32->bf16.
        #pragma unroll
        for (int p = 0; p < 4; ++p) {
            f32x4 v = {0.f, 0.f, 0.f, 0.f};
            if (asrc[p]) v = *(const f32x4*)(asrc[p] + k0);
            st_bf16x4(&lA[(p * 32 + arow) * 32 + acol], v);
        }
        // --- W tiles (gate & up): 64 rows x 32 cols each, fp32->bf16.
        #pragma unroll
        for (int p = 0; p < 2; ++p) {
            int q   = p * 256 + tid;    // 0..511 float4 slots
            int row = q >> 3;           // 0..63
            int c   = (q & 7) * 4;
            f32x4 vg = *(const f32x4*)(wg + (size_t)row * D_DIM + k0 + c);
            f32x4 vu = *(const f32x4*)(wu + (size_t)row * D_DIM + k0 + c);
            st_bf16x4(&lG[row * 32 + c], vg);
            st_bf16x4(&lU[row * 32 + c], vu);
        }
        __syncthreads();

        // --- A fragment (ISA bf16 A layout): lane<16 -> K{0..7,16..23}
        BfFrag a;
        {
            int m   = wave * 16 + (lane & 15);
            int klo = (lane >= 16) ? 8 : 0;
            a.q[0] = *(const f32x4*)&lA[m * 32 + klo];
            a.q[1] = *(const f32x4*)&lA[m * 32 + klo + 16];
        }
        const int nb = (lane & 15) * 32 + ((lane >= 16) ? 8 : 0);

        // Batch-load 4 gate B-fragments, then issue the WMMA chain (one LDS
        // wait covers 4 matrix ops instead of one wait per WMMA).
        BfFrag b[4];
        #pragma unroll
        for (int ns = 0; ns < 4; ++ns) {
            b[ns].q[0] = *(const f32x4*)&lG[ns * 16 * 32 + nb];
            b[ns].q[1] = *(const f32x4*)&lG[ns * 16 * 32 + nb + 16];
        }
        #pragma unroll
        for (int ns = 0; ns < 4; ++ns)
            accg[ns] = __builtin_amdgcn_wmma_f32_16x16x32_bf16(
                false, a.v, false, b[ns].v, (short)0, accg[ns], false, false);
        #pragma unroll
        for (int ns = 0; ns < 4; ++ns) {
            b[ns].q[0] = *(const f32x4*)&lU[ns * 16 * 32 + nb];
            b[ns].q[1] = *(const f32x4*)&lU[ns * 16 * 32 + nb + 16];
        }
        #pragma unroll
        for (int ns = 0; ns < 4; ++ns)
            accu[ns] = __builtin_amdgcn_wmma_f32_16x16x32_bf16(
                false, a.v, false, b[ns].v, (short)0, accu[ns], false, false);
    }

    // --- Epilogue: h = silu(gate) * up -> bf16 workspace
    const int mbase = mtile * 128 + wave * 16 + ((lane >= 16) ? 8 : 0);
    #pragma unroll
    for (int ns = 0; ns < 4; ++ns) {
        int n = ntile * 64 + ns * 16 + (lane & 15);
        #pragma unroll
        for (int r = 0; r < 8; ++r) {
            int g = mbase + r;
            if (g < cnt) {
                float gt = accg[ns][r];
                float up = accu[ns][r];
                float h  = (gt / (1.0f + __expf(-gt))) * up;
                hbuf[(size_t)(e * T_TOK + g) * I_DIM + n] = (__bf16)h;
            }
        }
    }
}

// ---------------------------------------------------------------------------
// Kernel 3: down GEMM, scale by routing weight, atomic-accumulate into out.
// A tile (already bf16 in workspace) is staged with CDNA5 async global->LDS
// DMA, overlapping with the fp32->bf16 conversion of the B tile.
// ---------------------------------------------------------------------------
__global__ __launch_bounds__(256) void moe_down_kernel(
    const __bf16* __restrict__ hbuf,   // (E*T, I) bf16
    const float*  __restrict__ w_d,    // (E, D, I) fp32
    const int*    __restrict__ counts,
    const int*    __restrict__ list,
    const float*  __restrict__ wlist,
    float*        __restrict__ out)    // (T, D) fp32
{
    const int e     = blockIdx.z;
    const int mtile = blockIdx.y;
    const int ntile = blockIdx.x;      // 64-col tiles over D (32 tiles)
    const int cnt   = counts[e];
    if (mtile * 128 >= cnt) return;

    __shared__ __align__(16) __bf16 lA[128 * 32];
    __shared__ __align__(16) __bf16 lB[64 * 32];

    const int tid  = threadIdx.x;
    const int lane = tid & 31;
    const int wave = tid >> 5;

    const float* wd = w_d + (size_t)e * D_DIM * I_DIM + (size_t)(ntile * 64) * I_DIM;

    // Per-thread A-tile slots (16B each, 2 passes of 64 rows).
    const int ar = tid >> 2;           // 0..63
    const int ac = (tid & 3) * 8;      // 8 bf16 = 16B
    const __bf16* asrc0 = hbuf + (size_t)(e * T_TOK + mtile * 128 + ar)      * I_DIM + ac;
    const __bf16* asrc1 = hbuf + (size_t)(e * T_TOK + mtile * 128 + 64 + ar) * I_DIM + ac;
    const unsigned lds_a0 = (unsigned)(size_t)&lA[ar        * 32 + ac];
    const unsigned lds_a1 = (unsigned)(size_t)&lA[(64 + ar) * 32 + ac];

    v8f acc[4] = {};

    for (int k0 = 0; k0 < I_DIM; k0 += 32) {
        __syncthreads();
        // --- A tile: async DMA global->LDS (no VGPR round trip).
        async_copy_b128(lds_a0, asrc0 + k0);
        async_copy_b128(lds_a1, asrc1 + k0);
        // --- B tile: 64 rows x 32 cols of down_proj, fp32->bf16 (overlaps DMA).
        #pragma unroll
        for (int p = 0; p < 2; ++p) {
            int q   = p * 256 + tid;
            int row = q >> 3;
            int c   = (q & 7) * 4;
            f32x4 v = *(const f32x4*)(wd + (size_t)row * I_DIM + k0 + c);
            st_bf16x4(&lB[row * 32 + c], v);
        }
        wait_asynccnt0();
        __syncthreads();

        BfFrag a;
        {
            int m   = wave * 16 + (lane & 15);
            int klo = (lane >= 16) ? 8 : 0;
            a.q[0] = *(const f32x4*)&lA[m * 32 + klo];
            a.q[1] = *(const f32x4*)&lA[m * 32 + klo + 16];
        }
        const int nb = (lane & 15) * 32 + ((lane >= 16) ? 8 : 0);
        BfFrag b[4];
        #pragma unroll
        for (int ns = 0; ns < 4; ++ns) {
            b[ns].q[0] = *(const f32x4*)&lB[ns * 16 * 32 + nb];
            b[ns].q[1] = *(const f32x4*)&lB[ns * 16 * 32 + nb + 16];
        }
        #pragma unroll
        for (int ns = 0; ns < 4; ++ns)
            acc[ns] = __builtin_amdgcn_wmma_f32_16x16x32_bf16(
                false, a.v, false, b[ns].v, (short)0, acc[ns], false, false);
    }

    const int mbase = mtile * 128 + wave * 16 + ((lane >= 16) ? 8 : 0);
    #pragma unroll
    for (int ns = 0; ns < 4; ++ns) {
        int n = ntile * 64 + ns * 16 + (lane & 15);
        #pragma unroll
        for (int r = 0; r < 8; ++r) {
            int g = mbase + r;
            if (g < cnt) {
                int   tok = list[e * T_TOK + g];
                float w   = wlist[e * T_TOK + g];
                atomicAdd(out + (size_t)tok * D_DIM + n, acc[ns][r] * w);
            }
        }
    }
}

// ---------------------------------------------------------------------------
extern "C" void kernel_launch(void* const* d_in, const int* in_sizes, int n_in,
                              void* d_out, int out_size, void* d_ws, size_t ws_size,
                              hipStream_t stream) {
    const float* hs   = (const float*)d_in[0];   // hidden_states (T,D)
    const int*   topk = (const int*)d_in[1];     // top_k_index (T,K)
    const float* topw = (const float*)d_in[2];   // top_k_weights (T,K)
    const float* w_gu = (const float*)d_in[3];   // gate_up_proj (E,2I,D)
    const float* w_d  = (const float*)d_in[4];   // down_proj (E,D,I)
    float* out = (float*)d_out;

    // Workspace layout:
    //   [0,   64)          counts (E ints)
    //   [256, 256+128K)    list   (E*T ints)
    //   [..., ...)         wlist  (E*T floats)
    //   [512K, 512K+~88MB) hbuf   (E*T x I bf16)
    char* ws = (char*)d_ws;
    int*    counts = (int*)ws;
    int*    list   = (int*)(ws + 256);
    float*  wlist  = (float*)(ws + 256 + (size_t)E_EXP * T_TOK * 4);
    __bf16* hbuf   = (__bf16*)(ws + (size_t)512 * 1024);

    const int n4 = (T_TOK * D_DIM) / 4;
    moe_zero_kernel<<<(n4 + 255) / 256, 256, 0, stream>>>(out, n4, counts);
    moe_route_kernel<<<1, 32, 0, stream>>>(topk, topw, counts, list, wlist);
    moe_gateup_kernel<<<dim3(I_DIM / 64, T_TOK / 128, E_EXP), 256, 0, stream>>>(
        hs, w_gu, counts, list, hbuf);
    moe_down_kernel<<<dim3(D_DIM / 64, T_TOK / 128, E_EXP), 256, 0, stream>>>(
        hbuf, w_d, counts, list, wlist, out);
}
